// MultiHeadedAttention_36971078484381
// MI455X (gfx1250) — matrix operand order repaired and checked by reading, not compiled, not run on hardware
//
#include <hip/hip_runtime.h>

// MultiHeadedAttention with banded (restrict=31) local attention, CDNA5 WMMA.
// B=4, T=2048, D_MODEL=512, H=8, D_K=64. bf16 WMMA (16x16x32), f32 accumulate.
// K/V workspaces are zero-padded (16 slots front, 48 back) so the attention
// kernel is branch-free: no EXEC-masked regions around any WMMA.

#define B_    4
#define T_    2048
#define H_    8
#define DM    512
#define DK    64
#define RWIN  31
#define RHALF 15
#define KROWS (T_ + 64)          // padded token slots: pos -16 .. T+47

typedef __attribute__((ext_vector_type(16))) __bf16 v16bf;
typedef __attribute__((ext_vector_type(8)))  __bf16 v8bf;
typedef __attribute__((ext_vector_type(8)))  float  v8f;
typedef __attribute__((ext_vector_type(4)))  float  v4f;
typedef unsigned short ushort_t;

__device__ __forceinline__ __bf16 to_bf16(float f) { return (__bf16)f; }

__device__ __forceinline__ v16bf cat8(v8bf lo, v8bf hi) {
    return __builtin_shufflevector(lo, hi, 0,1,2,3,4,5,6,7,8,9,10,11,12,13,14,15);
}

// ---------------------------------------------------------------------------
// f32 -> bf16 elementwise (8 elems / thread, vector loads/stores)
// ---------------------------------------------------------------------------
__global__ void cvt_f32_bf16(const float* __restrict__ in,
                             ushort_t* __restrict__ out, int n8)
{
    const int i = blockIdx.x * blockDim.x + threadIdx.x;
    if (i >= n8) return;
    const v4f* p = (const v4f*)(in + (size_t)i * 8);
    v4f a = p[0], b = p[1];
    v8bf o;
#pragma unroll
    for (int j = 0; j < 4; ++j) { o[j] = to_bf16(a[j]); o[j + 4] = to_bf16(b[j]); }
    *(v8bf*)(out + (size_t)i * 8) = o;
}

// ---------------------------------------------------------------------------
// Zero the pad slots of Khp (bh, KROWS, DK) and Vtp (bh, DK, KROWS).
// Rows/positions [0,16) and [T+16, T+64) are pads. 8 bf16 per thread.
// ---------------------------------------------------------------------------
__global__ void zero_pads(ushort_t* __restrict__ Khp, ushort_t* __restrict__ Vtp)
{
    const int c = blockIdx.x * blockDim.x + threadIdx.x;   // 32768 chunks
    v8bf z = {};
    if (c < 16384) {                       // K pads: 32 bh x 64 rows x 8 chunks
        const int bh = c >> 9;
        const int r  = (c & 511) >> 3;
        const int ch = c & 7;
        const int row = (r < 16) ? r : (T_ + r);           // [0,16) U [T+16,T+64)
        *(v8bf*)(Khp + ((size_t)bh * KROWS + row) * DK + ch * 8) = z;
    } else {                               // V pads: 32 bh x 64 dk x 8 chunks
        const int c2 = c - 16384;
        const int bh = c2 >> 9;
        const int dk = (c2 & 511) >> 3;
        const int ch = c2 & 7;
        const int pos = (ch < 2) ? (ch * 8) : (T_ + ch * 8);
        *(v8bf*)(Vtp + ((size_t)bh * DK + dk) * KROWS + pos) = z;
    }
}

// ---------------------------------------------------------------------------
// Swizzle W (DMxDM f32, row-major) into WMMA B-fragment order, bf16:
// element (k,n) -> dest[(( (k>>5)*(DM/16) + (n>>4) )*32 + lane)*16 + (k&15)]
// with lane = (n&15) + 16*((k>>4)&1). One 32B contiguous load per lane/frag.
// ---------------------------------------------------------------------------
__global__ void swizzle_w_bf16(const float* __restrict__ W,
                               ushort_t* __restrict__ Wb)
{
    const int e = blockIdx.x * blockDim.x + threadIdx.x;   // DM*DM elements
    const int k = e >> 9;
    const int n = e & (DM - 1);
    const int lane = (n & 15) + 16 * ((k >> 4) & 1);
    const size_t dst = (((size_t)(k >> 5) * (DM >> 4) + (n >> 4)) * 32 + lane) * 16
                     + (k & 15);
    __bf16 v = to_bf16(W[e]);
    Wb[dst] = __builtin_bit_cast(ushort_t, v);
}

// ---------------------------------------------------------------------------
// GEMM: out = A(8192x512 bf16 row-major) * Wb(512x512 swizzled bf16) + bias.
// One wave computes a 32x32 tile (2 m-blocks x 2 n-tiles, 2x A and B reuse).
// Compile-time dims -> fully unrolled K loop, immediate-offset b128 loads.
// MODE 0: f32 out[m*DM + n]                              (final projection)
// MODE 1: bf16 Q head-major   out[(bh*T + t)*DK + dk]
// MODE 2: bf16 V transposed   out[(bh*DK + dk)*KROWS + 16 + t]  (16B stores)
// MODE 3: bf16 K padded       out[(bh*KROWS + 16 + t)*DK + dk]
// ---------------------------------------------------------------------------
template <int MODE>
__global__ void wmma_gemm_bf16(const ushort_t* __restrict__ A,
                               const ushort_t* __restrict__ Wb,
                               const float* __restrict__ bias,
                               void* __restrict__ outp)
{
    const int lane  = threadIdx.x & 31;
    const int wave  = threadIdx.x >> 5;
    const int row16 = lane & 15;
    const int hi    = lane >> 4;

    const int m0  = blockIdx.x * 32;
    const int nt0 = (blockIdx.y * 4 + wave) * 2;     // first of 2 n-tiles

    const __bf16* arow0 = (const __bf16*)A + (size_t)(m0 + row16) * DM;
    const __bf16* arow1 = arow0 + (size_t)16 * DM;
    const __bf16* wbase = (const __bf16*)Wb + ((size_t)nt0 * 32 + lane) * 16;

    v8f acc00 = {}, acc01 = {}, acc10 = {}, acc11 = {};
#pragma unroll
    for (int ks = 0; ks < DM / 32; ++ks) {
        const int kb = ks * 32;
        v16bf a0 = cat8(*(const v8bf*)(arow0 + kb + 16 * hi),
                        *(const v8bf*)(arow0 + kb + 8 + 16 * hi));
        v16bf a1 = cat8(*(const v8bf*)(arow1 + kb + 16 * hi),
                        *(const v8bf*)(arow1 + kb + 8 + 16 * hi));
        const __bf16* fb = wbase + (size_t)ks * (DM >> 4) * 32 * 16;
        v16bf b0 = *(const v16bf*)fb;
        v16bf b1 = *(const v16bf*)(fb + 32 * 16);
        acc00 = __builtin_amdgcn_wmma_f32_16x16x32_bf16(
                    false, a0, false, b0, (short)0, acc00, false, false);
        acc01 = __builtin_amdgcn_wmma_f32_16x16x32_bf16(
                    false, a0, false, b1, (short)0, acc01, false, false);
        acc10 = __builtin_amdgcn_wmma_f32_16x16x32_bf16(
                    false, a1, false, b0, (short)0, acc10, false, false);
        acc11 = __builtin_amdgcn_wmma_f32_16x16x32_bf16(
                    false, a1, false, b1, (short)0, acc11, false, false);
    }

#pragma unroll
    for (int mb = 0; mb < 2; ++mb) {
#pragma unroll
        for (int u = 0; u < 2; ++u) {
            const v8f acc = (mb == 0) ? (u == 0 ? acc00 : acc01)
                                      : (u == 0 ? acc10 : acc11);
            const int n  = (nt0 + u) * 16 + row16;
            const int mB = m0 + mb * 16;
            const float bv = bias[n];
            if (MODE == 0) {
                float* out = (float*)outp;
#pragma unroll
                for (int j = 0; j < 8; ++j) {
                    const int m = mB + hi * 8 + j;
                    out[(size_t)m * DM + n] = acc[j] + bv;
                }
            } else if (MODE == 1) {
                __bf16* out = (__bf16*)outp;
                const int h = n >> 6, dk = n & (DK - 1);
#pragma unroll
                for (int j = 0; j < 8; ++j) {
                    const int m = mB + hi * 8 + j;
                    const int b = m >> 11, t = m & (T_ - 1);
                    out[(((size_t)(b * H_ + h)) * T_ + t) * DK + dk] =
                        to_bf16(acc[j] + bv);
                }
            } else if (MODE == 3) {
                __bf16* out = (__bf16*)outp;
                const int h = n >> 6, dk = n & (DK - 1);
#pragma unroll
                for (int j = 0; j < 8; ++j) {
                    const int m = mB + hi * 8 + j;
                    const int b = m >> 11, t = m & (T_ - 1);
                    out[(((size_t)(b * H_ + h)) * KROWS + 16 + t) * DK + dk] =
                        to_bf16(acc[j] + bv);
                }
            } else {   // MODE 2
                __bf16* out = (__bf16*)outp;
                const int h = n >> 6, dk = n & (DK - 1);
                const int b = mB >> 11, tb = (mB & (T_ - 1)) + hi * 8;
                v8bf pk;
#pragma unroll
                for (int j = 0; j < 8; ++j) pk[j] = to_bf16(acc[j] + bv);
                *(v8bf*)(out + (((size_t)(b * H_ + h)) * DK + dk) * KROWS
                         + 16 + tb) = pk;
            }
        }
    }
}

// ---------------------------------------------------------------------------
// Banded local attention, one wave per (b, h, 16-token tile). Branch-free.
// Qh: (B*H, T, DK) bf16.  Khp: (B*H, KROWS, DK) bf16, row = 16 + pos.
// Vtp: (B*H, DK, KROWS) bf16, col = 16 + pos.  Xa: (B*T, DM) bf16.
// 64 key slots, pos = t0-16+slot; pads are zero -> score/value exactly 0.
// ---------------------------------------------------------------------------
__global__ void local_attn_wmma(const ushort_t* __restrict__ Qh,
                                const ushort_t* __restrict__ Khp,
                                const ushort_t* __restrict__ Vtp,
                                const int*      __restrict__ mask,
                                ushort_t* __restrict__ Xa)
{
    __shared__ __bf16 s_tile[8][16 * 64];   // per-wave 16x64 staging (16 KB)

    const int lane  = threadIdx.x & 31;
    const int wave  = threadIdx.x >> 5;
    const int row16 = lane & 15;
    const int hi    = lane >> 4;

    const int wglobal = blockIdx.x * 8 + wave;
    const int bh = wglobal >> 7;            // / (T/16 = 128)
    const int t0 = (wglobal & 127) * 16;
    const int b  = bh >> 3;
    const int h  = bh & 7;
    const int base = t0 - 16;               // key-slot 0 position

    const __bf16* qbase = (const __bf16*)Qh  + (size_t)bh * T_ * DK;
    const __bf16* kbase = (const __bf16*)Khp + ((size_t)bh * KROWS + 16) * DK;
    const __bf16* vbase = (const __bf16*)Vtp + (size_t)bh * DK * KROWS + 16;

    // ---- Q A-fragments: rows t0..t0+15, K=64 as two 32-chunks ----
    v16bf qa[2];
    {
        const __bf16* qr = qbase + (size_t)(t0 + row16) * DK;
#pragma unroll
        for (int s = 0; s < 2; ++s)
            qa[s] = cat8(*(const v8bf*)(qr + s * 32 + 16 * hi),
                         *(const v8bf*)(qr + s * 32 + 8 + 16 * hi));
    }

    // ---- Scores: 4 slot-tiles x 2 k-steps = 8 WMMAs (no predication) ----
    v8f sc[4];
#pragma unroll
    for (int nt = 0; nt < 4; ++nt) {
        v8f z = {};
        sc[nt] = z;
        const int pos = base + nt * 16 + row16;   // in [-16, T+47], padded
        const __bf16* krow = kbase + (size_t)pos * DK;
#pragma unroll
        for (int s = 0; s < 2; ++s) {
            v16bf kf = *(const v16bf*)(krow + s * 32 + hi * 16);
            sc[nt] = __builtin_amdgcn_wmma_f32_16x16x32_bf16(
                         false, qa[s], false, kf, (short)0, sc[nt], false, false);
        }
    }

    // ---- Per-row softmax over the 31-entry band (half-wave reductions) ----
    const float scale = 0.125f;             // 1/sqrt(64)
    float attnv[4][8];
#pragma unroll
    for (int j = 0; j < 8; ++j) {
        const int t = t0 + hi * 8 + j;      // this half-wave's row
        float sv[4];
        bool  incl[4];
        float mx = -3.0e38f;
#pragma unroll
        for (int nt = 0; nt < 4; ++nt) {
            const int pos  = base + nt * 16 + row16;
            const int jrel = pos - t + RHALF;
            incl[nt] = (jrel >= 0) && (jrel < RWIN);
            sv[nt]   = sc[nt][j] * scale;
            if (incl[nt] && sv[nt] > mx) mx = sv[nt];
        }
#pragma unroll
        for (int d = 1; d < 16; d <<= 1)
            mx = fmaxf(mx, __shfl_xor(mx, d, 32));
        float lsum = 0.0f;
        float e[4];
#pragma unroll
        for (int nt = 0; nt < 4; ++nt) {
            e[nt] = incl[nt] ? __expf(sv[nt] - mx) : 0.0f;
            lsum += e[nt];
        }
#pragma unroll
        for (int d = 1; d < 16; d <<= 1)
            lsum += __shfl_xor(lsum, d, 32);
        const float inv = 1.0f / lsum;      // band always has 31 entries
#pragma unroll
        for (int nt = 0; nt < 4; ++nt)
            attnv[nt][j] = e[nt] * inv;
    }

    // ---- Relayout attn C->A via private LDS (wave-local, no barrier) ----
    __bf16* sm = &s_tile[wave][0];
#pragma unroll
    for (int j = 0; j < 8; ++j) {
        const int m = hi * 8 + j;
#pragma unroll
        for (int nt = 0; nt < 4; ++nt)
            sm[m * 64 + nt * 16 + row16] = to_bf16(attnv[nt][j]);
    }
    v16bf aat[2];
#pragma unroll
    for (int s = 0; s < 2; ++s)
        aat[s] = cat8(*(const v8bf*)(sm + row16 * 64 + s * 32 + 16 * hi),
                      *(const v8bf*)(sm + row16 * 64 + s * 32 + 8 + 16 * hi));

    // ---- X = attn (16x64) x Vblock (64x64): 8 WMMAs (no predication) ----
    v8f xacc[4];
#pragma unroll
    for (int vt = 0; vt < 4; ++vt) { v8f z = {}; xacc[vt] = z; }
#pragma unroll
    for (int s = 0; s < 2; ++s) {
#pragma unroll
        for (int vt = 0; vt < 4; ++vt) {
            const __bf16* vcol = vbase + (size_t)(vt * 16 + row16) * KROWS;
            v16bf vfv = *(const v16bf*)(vcol + base + s * 32 + hi * 16);
            xacc[vt] = __builtin_amdgcn_wmma_f32_16x16x32_bf16(
                           false, aat[s], false, vfv, (short)0, xacc[vt],
                           false, false);
        }
    }

    // ---- mask multiply, stage in LDS, then coalesced 32B stores ----
#pragma unroll
    for (int j = 0; j < 8; ++j) {
        const int m = hi * 8 + j;
        const float mk = (mask[(size_t)b * T_ + t0 + m] != 0) ? 1.0f : 0.0f;
#pragma unroll
        for (int vt = 0; vt < 4; ++vt)
            sm[m * 64 + vt * 16 + row16] = to_bf16(xacc[vt][j] * mk);
    }
    {
        const int m2 = lane >> 1, coff = (lane & 1) * 32;
        const v16bf* src = (const v16bf*)(sm + m2 * 64 + coff);
        v16bf x0 = src[0], x1 = src[1];
        __bf16* dst = (__bf16*)Xa + ((size_t)(b * T_ + t0 + m2)) * DM
                    + h * DK + coff;
        *(v16bf*)dst = x0;
        *((v16bf*)dst + 1) = x1;
    }
}

// ---------------------------------------------------------------------------
extern "C" void kernel_launch(void* const* d_in, const int* in_sizes, int n_in,
                              void* d_out, int out_size, void* d_ws, size_t ws_size,
                              hipStream_t stream)
{
    (void)in_sizes; (void)n_in; (void)out_size; (void)ws_size;

    const float* query = (const float*)d_in[0];
    const float* key   = (const float*)d_in[1];
    const float* value = (const float*)d_in[2];
    const int*   mask  = (const int*)  d_in[3];
    // d_in[4] = restrict (31, fixed window)
    const float* Wq = (const float*)d_in[5];
    const float* bq = (const float*)d_in[6];
    const float* Wk = (const float*)d_in[7];
    const float* bk = (const float*)d_in[8];
    const float* Wv = (const float*)d_in[9];
    const float* bv = (const float*)d_in[10];
    const float* Wo = (const float*)d_in[11];
    const float* bo = (const float*)d_in[12];

    const int M = B_ * T_;                     // 8192 tokens
    char* ws = (char*)d_ws;
    auto take = [&](size_t bytes) { char* p = ws; ws += bytes; return p; };

    ushort_t* Xq  = (ushort_t*)take((size_t)M * DM * 2);   // bf16 inputs
    ushort_t* Xk  = (ushort_t*)take((size_t)M * DM * 2);
    ushort_t* Xv  = (ushort_t*)take((size_t)M * DM * 2);
    ushort_t* Wbq = (ushort_t*)take((size_t)DM * DM * 2);  // swizzled weights
    ushort_t* Wbk = (ushort_t*)take((size_t)DM * DM * 2);
    ushort_t* Wbv = (ushort_t*)take((size_t)DM * DM * 2);
    ushort_t* Wbo = (ushort_t*)take((size_t)DM * DM * 2);
    ushort_t* Qh  = (ushort_t*)take((size_t)M * DM * 2);            // (B*H,T,DK)
    ushort_t* Khp = (ushort_t*)take((size_t)B_ * H_ * KROWS * DK * 2); // padded
    ushort_t* Vtp = (ushort_t*)take((size_t)B_ * H_ * DK * KROWS * 2); // padded
    ushort_t* Xa  = (ushort_t*)take((size_t)M * DM * 2);            // (B*T,DM)

    // pads + bf16 conversions + weight swizzles (one pass, stays in L2)
    {
        dim3 bl(256);
        zero_pads<<<dim3(32768 / 256), bl, 0, stream>>>(Khp, Vtp);
        const int n8 = (M * DM) / 8;           // 524288
        dim3 g(n8 / 256);
        cvt_f32_bf16<<<g, bl, 0, stream>>>(query, Xq, n8);
        cvt_f32_bf16<<<g, bl, 0, stream>>>(key,   Xk, n8);
        cvt_f32_bf16<<<g, bl, 0, stream>>>(value, Xv, n8);
        dim3 gw((DM * DM) / 256);
        swizzle_w_bf16<<<gw, bl, 0, stream>>>(Wq, Wbq);
        swizzle_w_bf16<<<gw, bl, 0, stream>>>(Wk, Wbk);
        swizzle_w_bf16<<<gw, bl, 0, stream>>>(Wv, Wbv);
        swizzle_w_bf16<<<gw, bl, 0, stream>>>(Wo, Wbo);
    }

    dim3 gg(M / 32, DM / 128);                 // 256 x 4 blocks
    dim3 bb(128);                              // 4 waves, 32x32 tile each

    wmma_gemm_bf16<1><<<gg, bb, 0, stream>>>(Xq, Wbq, bq, Qh);
    wmma_gemm_bf16<3><<<gg, bb, 0, stream>>>(Xk, Wbk, bk, Khp);
    wmma_gemm_bf16<2><<<gg, bb, 0, stream>>>(Xv, Wbv, bv, Vtp);

    const int totalWaves = B_ * H_ * (T_ / 16);    // 4096
    local_attn_wmma<<<totalWaves / 8, 256, 0, stream>>>(Qh, Khp, Vtp, mask, Xa);

    wmma_gemm_bf16<0><<<gg, bb, 0, stream>>>(Xa, Wbo, bo, d_out);
}
